// MVDeformAttn_16131897164038
// MI455X (gfx1250) — matrix operand order repaired and checked by reading, not compiled, not run on hardware
//
#include <hip/hip_runtime.h>
#include <hip/hip_bf16.h>
#include <stdint.h>

typedef __attribute__((ext_vector_type(16))) _Float16 v16h;
typedef __attribute__((ext_vector_type(8)))  _Float16 h8;
typedef __attribute__((ext_vector_type(8)))  float    v8f;

#define NB    2
#define LQ    2048
#define CDIM  256
#define MH    8
#define DH    32
#define LL    4
#define TC    2
#define PP    8
#define STOT  21760   // 128*128 + 64*64 + 32*32 + 16*16
#define BN_EPS 1e-5f

// ---------------------------------------------------------------------------
// Tensor Data Mover: 2D f16 tile (rows x elems_per_row, contiguous) -> LDS.
// D# per CDNA5 ISA. Issued by one wave; TDM
// ignores EXEC. Caller waits s_wait_tensorcnt then barriers.
// ---------------------------------------------------------------------------
#if __has_builtin(__builtin_amdgcn_tensor_load_to_lds) && \
    __has_builtin(__builtin_amdgcn_s_wait_tensorcnt)
#define HAVE_TDM 1
typedef unsigned int u32x4 __attribute__((ext_vector_type(4)));
typedef int i32x4 __attribute__((ext_vector_type(4)));
typedef int i32x8 __attribute__((ext_vector_type(8)));

__device__ __forceinline__ void tdm_load_2d_f16(void* lds_dst, const void* gsrc,
                                                int elems_per_row, int rows) {
  uint64_t ga = (uint64_t)(uintptr_t)gsrc;
  u32x4 g0;
  g0[0] = 1u;                                        // count=1, user descriptor
  g0[1] = (uint32_t)(uintptr_t)lds_dst;              // lds_addr (bytes)
  g0[2] = (uint32_t)ga;                              // global_addr[31:0]
  g0[3] = (uint32_t)((ga >> 32) & 0x01FFFFFFu) | (2u << 30);  // [56:32] + type=2
  i32x8 g1;
  g1[0] = 1 << 16;                                   // data_size = 2 bytes
  g1[1] = (elems_per_row & 0xFFFF) << 16;            // tensor_dim0[15:0]
  g1[2] = ((elems_per_row >> 16) & 0xFFFF) |
          ((rows & 0xFFFF) << 16);                   // dim0[31:16] | dim1[15:0]
  g1[3] = ((rows >> 16) & 0xFFFF) |
          ((elems_per_row & 0xFFFF) << 16);          // dim1[31:16] | tile_dim0
  g1[4] = rows & 0xFFFF;                             // tile_dim1 (tile_dim2=0)
  g1[5] = elems_per_row;                             // tensor_dim0_stride[31:0]
  g1[6] = 0;
  g1[7] = 0;
  i32x4 z4 = {0, 0, 0, 0};
#if __clang_major__ >= 23
  i32x8 z8 = {0, 0, 0, 0, 0, 0, 0, 0};
  __builtin_amdgcn_tensor_load_to_lds(g0, g1, z4, z4, z8, 0);
#else
  __builtin_amdgcn_tensor_load_to_lds(g0, g1, z4, z4, 0);
#endif
}
#endif

// ---------------------------------------------------------------------------
// WMMA fragment loaders (CDNA5 16x16x32 f16 layouts, CDNA5 ISA)
// ---------------------------------------------------------------------------
__device__ __forceinline__ v16h ld_a(const _Float16* tile, int row0, int k0,
                                     int ld, int lane) {
  union { v16h h; uint32_t u[8]; } r;
  const _Float16* base = tile + (row0 + (lane & 15)) * ld + k0;
  int kb = (lane >> 4) * 8;
  const uint32_t* p0 = (const uint32_t*)(base + kb);
  const uint32_t* p1 = (const uint32_t*)(base + 16 + kb);
#pragma unroll
  for (int v = 0; v < 4; ++v) { r.u[v] = p0[v]; r.u[4 + v] = p1[v]; }
  return r.h;
}

__device__ __forceinline__ v16h ld_b(const _Float16* wt, int col0, int k0,
                                     int ldk, int lane) {
  union { v16h h; uint32_t u[8]; } r;
  const _Float16* base = wt + (col0 + (lane & 15)) * ldk + k0;
  int kb = (lane >> 4) * 8;
  const uint32_t* p0 = (const uint32_t*)(base + kb);
  const uint32_t* p1 = (const uint32_t*)(base + 16 + kb);
#pragma unroll
  for (int v = 0; v < 4; ++v) { r.u[v] = p0[v]; r.u[4 + v] = p1[v]; }
  return r.h;
}

#define WMMA_F16(A, B, C) \
  __builtin_amdgcn_wmma_f32_16x16x32_f16(false, (A), false, (B), (short)0, (C), false, false)

// ---------------------------------------------------------------------------
// Prep: transpose+convert f32 weights [b][K][Nc] -> f16 [b][Nc][K]
// ---------------------------------------------------------------------------
__global__ void k_wt_transpose(const float* __restrict__ src,
                               _Float16* __restrict__ dst,
                               int K, int Nc, int total) {
  int i = blockIdx.x * blockDim.x + threadIdx.x;
  if (i >= total) return;
  int mat = K * Nc;
  int b = i / mat, r = i - b * mat;
  int k = r / Nc, c = r - k * Nc;
  dst[b * mat + c * K + k] = (_Float16)src[i];
}

// BN fold: scale = g*rsqrt(rv+eps); bias = b - rm*scale
__global__ void k_bnprep(const float* __restrict__ g, const float* __restrict__ b,
                         const float* __restrict__ rm, const float* __restrict__ rv,
                         float* __restrict__ sc, float* __restrict__ bi, int n) {
  int i = blockIdx.x * blockDim.x + threadIdx.x;
  if (i >= n) return;
  float s = g[i] * rsqrtf(rv[i] + BN_EPS);
  sc[i] = s;
  bi[i] = b[i] - rm[i] * s;
}

// ---------------------------------------------------------------------------
// Attention weights: softmax_P( query @ aw_w + aw_b )   [N*Lq*M threads]
// ---------------------------------------------------------------------------
__global__ void k_attn(const float* __restrict__ query, const float* __restrict__ aw,
                       const float* __restrict__ ab, float* __restrict__ attn) {
  int g = blockIdx.x * blockDim.x + threadIdx.x;  // (n*Lq+q)*8 + m
  int row = g >> 3, m = g & 7;
  const float* qp = query + row * CDIM;
  float lg[PP];
#pragma unroll
  for (int p = 0; p < PP; ++p) lg[p] = ab[m * PP + p];
  for (int k = 0; k < CDIM; ++k) {
    float qv = qp[k];
    const float* wr = aw + k * (MH * PP) + m * PP;
#pragma unroll
    for (int p = 0; p < PP; ++p) lg[p] += qv * wr[p];
  }
  float mx = lg[0];
#pragma unroll
  for (int p = 1; p < PP; ++p) mx = fmaxf(mx, lg[p]);
  float s = 0.f;
#pragma unroll
  for (int p = 0; p < PP; ++p) { lg[p] = __expf(lg[p] - mx); s += lg[p]; }
  float inv = 1.0f / s;
#pragma unroll
  for (int p = 0; p < PP; ++p) attn[g * PP + p] = lg[p] * inv;
}

// ---------------------------------------------------------------------------
// Value projection GEMM (WMMA): val = value @ vp_w + vp_b, scattered to
// f16 layout [level][n][m][t][hc][w][d]. Per-row scatter bases precomputed
// into LDS so the WMMA epilogue has no division / level-decode branches.
// ---------------------------------------------------------------------------
__global__ void __launch_bounds__(256)
k_value_proj(const float* __restrict__ value, const _Float16* __restrict__ vpwt,
             const float* __restrict__ vpb, _Float16* __restrict__ valh) {
  __shared__ __attribute__((aligned(16))) _Float16 at[64 * 256];
  __shared__ long rbase[64];
  __shared__ int  mstep[64];
  int tid = threadIdx.x;
  int row0 = blockIdx.x * 64;
  const float* src = value + row0 * CDIM;
  for (int i = tid; i < 64 * 256; i += 256) at[i] = (_Float16)src[i];
  if (tid < 64) {  // per-row scatter base: dst = rbase + m*mstep + d
    int grow = row0 + tid;
    int n = grow / STOT, s = grow - n * STOT;
    int bs, lwv, lhv; long lb;
    if (s < 16384)      { bs = 0;     lwv = 7; lhv = 6; lb = 0;        }
    else if (s < 20480) { bs = 16384; lwv = 6; lhv = 5; lb = 8388608;  }
    else if (s < 21504) { bs = 20480; lwv = 5; lhv = 4; lb = 10485760; }
    else                { bs = 21504; lwv = 4; lhv = 3; lb = 11010048; }
    int sl = s - bs;
    int w  = sl & ((1 << lwv) - 1);
    int rr = sl >> lwv;
    int hc = rr & ((1 << lhv) - 1);
    int t  = rr >> lhv;
    int lsz = TC << (lhv + lwv + 5);           // elements per head per level
    long pix = ((long)((t << lhv) + hc) << lwv) + w;
    rbase[tid] = lb + (long)n * MH * lsz + (pix << 5);
    mstep[tid] = lsz;
  }
  __syncthreads();
  int lane = tid & 31, wv = tid >> 5;
  int hi = lane >> 4, ln = lane & 15;
  for (int cbi = 0; cbi < 2; ++cbi) {
    int cb = wv + cbi * 8;
    v16h bf[8];
#pragma unroll
    for (int ks = 0; ks < 8; ++ks) bf[ks] = ld_b(vpwt, cb * 16, ks * 32, 256, lane);
    int col = cb * 16 + ln;
    float bias = vpb[col];
    int m = col >> 5, d = col & 31;
    for (int rb = 0; rb < 4; ++rb) {
      v8f acc = {};
#pragma unroll
      for (int ks = 0; ks < 8; ++ks) {
        v16h af = ld_a(at, rb * 16, ks * 32, 256, lane);
        acc = WMMA_F16(af, bf[ks], acc);
      }
#pragma unroll
      for (int r = 0; r < 8; ++r) {
        int rl = rb * 16 + hi * 8 + r;
        valh[rbase[rl] + (long)m * mstep[rl] + d] = (_Float16)(acc[r] + bias);
      }
    }
  }
}

// ---------------------------------------------------------------------------
// Fused bilinear grid-sample + per-level stereo MLP (64->64->32, BN+ReLU).
// 64 rows (= 8 queries x 8 points) per workgroup; grid = (4096, L).
// ---------------------------------------------------------------------------
__global__ void __launch_bounds__(256)
k_sample_mlp(const _Float16* __restrict__ valh, const float* __restrict__ sloc,
             const _Float16* __restrict__ m1t, const _Float16* __restrict__ m2t,
             const float* __restrict__ m1sc, const float* __restrict__ m1bi,
             const float* __restrict__ m2sc, const float* __restrict__ m2bi,
             _Float16* __restrict__ feat) {
  __shared__ __attribute__((aligned(16))) _Float16 st[64 * 64];
  __shared__ __attribute__((aligned(16))) _Float16 ht[64 * 64];
  int tid = threadIdx.x;
  int l = blockIdx.y;
  long row0 = (long)blockIdx.x * 64;
  int lwv, lhv; long lb;
  if (l == 0)      { lwv = 7; lhv = 6; lb = 0;        }
  else if (l == 1) { lwv = 6; lhv = 5; lb = 8388608;  }
  else if (l == 2) { lwv = 5; lhv = 4; lb = 10485760; }
  else             { lwv = 4; lhv = 3; lb = 11010048; }
  int Wl = 1 << lwv, Hcl = 1 << lhv;

  { // sampling: 128 tasks (row_local, cam), 16 channels per thread
    int task = tid >> 1;
    int dh = (tid & 1) * 16;
    int rl = task >> 1;
    int t  = task & 1;
    long grow = row0 + rl;
    int p = (int)(grow & 7);
    int q = (int)((grow >> 3) & (LQ - 1));
    int m = (int)((grow >> 14) & 7);
    int n = (int)(grow >> 17);
    long sidx = (((((long)(n * LQ + q) * MH + m) * LL + l) * TC + t) * PP + p) * 2;
    // normalize->grid->unnormalize collapses to (loc - 0.5)
    float x = sloc[sidx] - 0.5f;
    float y = sloc[sidx + 1] - 0.5f;
    float xf = floorf(x), yf = floorf(y);
    int x0 = (int)xf, y0 = (int)yf;
    float wx1 = x - xf, wy1 = y - yf;
    float wx0 = 1.f - wx1, wy0 = 1.f - wy1;
    long vb = lb + ((long)((n * MH + m) * TC + t) << (lhv + lwv + 5));
    float av[16];
#pragma unroll
    for (int i = 0; i < 16; ++i) av[i] = 0.f;
#pragma unroll
    for (int cy = 0; cy < 2; ++cy) {
      int yy = y0 + cy;
      if (yy < 0 || yy >= Hcl) continue;
      float wy = cy ? wy1 : wy0;
#pragma unroll
      for (int cx = 0; cx < 2; ++cx) {
        int xx = x0 + cx;
        if (xx < 0 || xx >= Wl) continue;
        float wxy = (cx ? wx1 : wx0) * wy;
        const h8* pv = (const h8*)(valh + vb + ((((long)yy << lwv) + xx) << 5) + dh);
        h8 a = pv[0], b = pv[1];
#pragma unroll
        for (int i = 0; i < 8; ++i) {
          av[i]     += wxy * (float)a[i];
          av[8 + i] += wxy * (float)b[i];
        }
      }
    }
    h8 o0, o1;
#pragma unroll
    for (int i = 0; i < 8; ++i) { o0[i] = (_Float16)av[i]; o1[i] = (_Float16)av[8 + i]; }
    h8* op = (h8*)(&st[rl * 64 + t * 32 + dh]);
    op[0] = o0; op[1] = o1;
  }
  __syncthreads();

  int lane = tid & 31, wv = tid >> 5;
  int hi = lane >> 4, ln = lane & 15;

  // MLP1: s(64x64) @ m1t -> relu(bn) -> ht(64x64); 16 tiles, 2 per wave
  const _Float16* m1w = m1t + l * 64 * 64;
  const float* sc1 = m1sc + l * 64;
  const float* bi1 = m1bi + l * 64;
#pragma unroll
  for (int ti = 0; ti < 2; ++ti) {
    int t = wv * 2 + ti;
    int rb = t >> 2, cb = t & 3;
    v8f acc = {};
#pragma unroll
    for (int ks = 0; ks < 2; ++ks) {
      v16h af = ld_a(st, rb * 16, ks * 32, 64, lane);
      v16h bf = ld_b(m1w, cb * 16, ks * 32, 64, lane);
      acc = WMMA_F16(af, bf, acc);
    }
    int col = cb * 16 + ln;
    float s = sc1[col], b = bi1[col];
#pragma unroll
    for (int r = 0; r < 8; ++r) {
      float x = fmaxf(acc[r] * s + b, 0.f);
      ht[(rb * 16 + hi * 8 + r) * 64 + col] = (_Float16)x;
    }
  }
  __syncthreads();

  // MLP2: ht(64x64) @ m2t -> relu(bn) -> feat[:, l*32 .. ]; 8 tiles, 1/wave
  const _Float16* m2w = m2t + l * 32 * 64;
  const float* sc2 = m2sc + l * 32;
  const float* bi2 = m2bi + l * 32;
  {
    int rb = wv >> 1, cb = wv & 1;
    v8f acc = {};
#pragma unroll
    for (int ks = 0; ks < 2; ++ks) {
      v16h af = ld_a(ht, rb * 16, ks * 32, 64, lane);
      v16h bf = ld_b(m2w, cb * 16, ks * 32, 64, lane);
      acc = WMMA_F16(af, bf, acc);
    }
    int col = cb * 16 + ln;
    float s = sc2[col], b = bi2[col];
#pragma unroll
    for (int r = 0; r < 8; ++r) {
      float x = fmaxf(acc[r] * s + b, 0.f);
      long grow = row0 + rb * 16 + hi * 8 + r;
      feat[grow * 128 + l * 32 + col] = (_Float16)x;
    }
  }
}

// ---------------------------------------------------------------------------
// Fusion MLP (128->128->32, BN+ReLU) + attention-weighted P reduction.
// A-tile (64x128 f16) staged into LDS by the Tensor Data Mover.
// ---------------------------------------------------------------------------
__global__ void __launch_bounds__(256)
k_fusion(const _Float16* __restrict__ feat,
         const _Float16* __restrict__ f1t, const _Float16* __restrict__ f2t,
         const float* __restrict__ f1sc, const float* __restrict__ f1bi,
         const float* __restrict__ f2sc, const float* __restrict__ f2bi,
         const float* __restrict__ attn, _Float16* __restrict__ pin) {
  __shared__ __attribute__((aligned(16))) _Float16 ft[64 * 128];
  __shared__ __attribute__((aligned(16))) _Float16 h1[64 * 128];
  int tid = threadIdx.x;
  long row0 = (long)blockIdx.x * 64;
#ifdef HAVE_TDM
  if ((tid >> 5) == 0) {                       // wave 0 drives the TDM
    tdm_load_2d_f16(ft, feat + row0 * 128, 128, 64);
    __builtin_amdgcn_s_wait_tensorcnt(0);
  }
#else
  {
    const uint32_t* src = (const uint32_t*)(feat + row0 * 128);
    uint32_t* dstf = (uint32_t*)ft;
    for (int i = tid; i < 64 * 128 / 2; i += 256) dstf[i] = src[i];
  }
#endif
  __syncthreads();
  int lane = tid & 31, wv = tid >> 5;
  int hi = lane >> 4, ln = lane & 15;

  { // h1 = relu(bn(ft @ f1)): wave wv owns col-block wv over 4 row blocks
    v16h bf[4];
#pragma unroll
    for (int ks = 0; ks < 4; ++ks) bf[ks] = ld_b(f1t, wv * 16, ks * 32, 128, lane);
    int col = wv * 16 + ln;
    float s = f1sc[col], b = f1bi[col];
    for (int rb = 0; rb < 4; ++rb) {
      v8f acc = {};
#pragma unroll
      for (int ks = 0; ks < 4; ++ks) {
        v16h af = ld_a(ft, rb * 16, ks * 32, 128, lane);
        acc = WMMA_F16(af, bf[ks], acc);
      }
#pragma unroll
      for (int r = 0; r < 8; ++r) {
        float x = fmaxf(acc[r] * s + b, 0.f);
        h1[(rb * 16 + hi * 8 + r) * 128 + col] = (_Float16)x;
      }
    }
  }
  __syncthreads();

  { // h2 = relu(bn(h1 @ f2)), then sum_p h2 * attn -> pin
    int rb = wv >> 1, cb = wv & 1;
    v8f acc = {};
#pragma unroll
    for (int ks = 0; ks < 4; ++ks) {
      v16h af = ld_a(h1, rb * 16, ks * 32, 128, lane);
      v16h bf = ld_b(f2t, cb * 16, ks * 32, 128, lane);
      acc = WMMA_F16(af, bf, acc);
    }
    int col = cb * 16 + ln;
    float s = f2sc[col], b = f2bi[col];
    long grp = (row0 >> 3) + rb * 2 + hi;   // [n][m][q] group
    int q = (int)(grp & (LQ - 1));
    int m = (int)((grp >> 11) & 7);
    int n = (int)(grp >> 14);
    const float* ap = attn + (((long)(n * LQ + q) * MH + m) * PP);
    float o = 0.f;
#pragma unroll
    for (int r = 0; r < 8; ++r) {
      float x = fmaxf(acc[r] * s + b, 0.f);
      o += x * ap[r];                        // value r <-> point p = r
    }
    pin[(long)(n * LQ + q) * 256 + m * 32 + col] = (_Float16)o;
  }
}

// ---------------------------------------------------------------------------
// Output projection (WMMA): out = pin @ op_w + op_b -> f32 d_out.
// A-tile (64x256 f16) staged into LDS by the Tensor Data Mover.
// ---------------------------------------------------------------------------
__global__ void __launch_bounds__(256)
k_out_proj(const _Float16* __restrict__ pin, const _Float16* __restrict__ opwt,
           const float* __restrict__ opb, float* __restrict__ out) {
  __shared__ __attribute__((aligned(16))) _Float16 at[64 * 256];
  int tid = threadIdx.x;
  int row0 = blockIdx.x * 64;
#ifdef HAVE_TDM
  if ((tid >> 5) == 0) {
    tdm_load_2d_f16(at, pin + (long)row0 * 256, 256, 64);
    __builtin_amdgcn_s_wait_tensorcnt(0);
  }
#else
  {
    const uint32_t* src = (const uint32_t*)(pin + (long)row0 * 256);
    uint32_t* d = (uint32_t*)at;
    for (int i = tid; i < 64 * 256 / 2; i += 256) d[i] = src[i];
  }
#endif
  __syncthreads();
  int lane = tid & 31, wv = tid >> 5;
  int hi = lane >> 4, ln = lane & 15;
  for (int cbi = 0; cbi < 2; ++cbi) {
    int cb = wv + cbi * 8;
    v16h bf[8];
#pragma unroll
    for (int ks = 0; ks < 8; ++ks) bf[ks] = ld_b(opwt, cb * 16, ks * 32, 256, lane);
    int col = cb * 16 + ln;
    float bias = opb[col];
    for (int rb = 0; rb < 4; ++rb) {
      v8f acc = {};
#pragma unroll
      for (int ks = 0; ks < 8; ++ks) {
        v16h af = ld_a(at, rb * 16, ks * 32, 256, lane);
        acc = WMMA_F16(af, bf[ks], acc);
      }
#pragma unroll
      for (int r = 0; r < 8; ++r)
        out[(long)(row0 + rb * 16 + hi * 8 + r) * 256 + col] = acc[r] + bias;
    }
  }
}

// ---------------------------------------------------------------------------
extern "C" void kernel_launch(void* const* d_in, const int* in_sizes, int n_in,
                              void* d_out, int out_size, void* d_ws, size_t ws_size,
                              hipStream_t stream) {
  const float* query = (const float*)d_in[0];
  const float* value = (const float*)d_in[1];
  // d_in[2] spatial_shapes (int) — shapes are compile-time constants here
  const float* sloc  = (const float*)d_in[3];
  const float* vp_w  = (const float*)d_in[4];
  const float* vp_b  = (const float*)d_in[5];
  const float* aw_w  = (const float*)d_in[6];
  const float* aw_b  = (const float*)d_in[7];
  const float* m1_w  = (const float*)d_in[8];
  const float* m1_g  = (const float*)d_in[9];
  const float* m1_b  = (const float*)d_in[10];
  const float* m1_rm = (const float*)d_in[11];
  const float* m1_rv = (const float*)d_in[12];
  const float* m2_w  = (const float*)d_in[13];
  const float* m2_g  = (const float*)d_in[14];
  const float* m2_b  = (const float*)d_in[15];
  const float* m2_rm = (const float*)d_in[16];
  const float* m2_rv = (const float*)d_in[17];
  const float* f1_w  = (const float*)d_in[18];
  const float* f1_g  = (const float*)d_in[19];
  const float* f1_b  = (const float*)d_in[20];
  const float* f1_rm = (const float*)d_in[21];
  const float* f1_rv = (const float*)d_in[22];
  const float* f2_w  = (const float*)d_in[23];
  const float* f2_g  = (const float*)d_in[24];
  const float* f2_b  = (const float*)d_in[25];
  const float* f2_rm = (const float*)d_in[26];
  const float* f2_rv = (const float*)d_in[27];
  const float* op_w  = (const float*)d_in[28];
  const float* op_b  = (const float*)d_in[29];
  float* out = (float*)d_out;

  char* ws = (char*)d_ws;
  _Float16* vp_wt = (_Float16*)(ws + 0);         // 256x256 f16      131072 B
  _Float16* op_wt = (_Float16*)(ws + 131072);    // 256x256 f16
  _Float16* m1t   = (_Float16*)(ws + 262144);    // 4x64x64 f16       32768 B
  _Float16* m2t   = (_Float16*)(ws + 294912);    // 4x32x64 f16       16384 B
  _Float16* f1t   = (_Float16*)(ws + 311296);    // 128x128 f16       32768 B
  _Float16* f2t   = (_Float16*)(ws + 344064);    // 32x128 f16         8192 B
  float* m1sc = (float*)(ws + 352256);
  float* m1bi = (float*)(ws + 353280);
  float* m2sc = (float*)(ws + 354304);
  float* m2bi = (float*)(ws + 354816);
  float* f1sc = (float*)(ws + 355328);
  float* f1bi = (float*)(ws + 355840);
  float* f2sc = (float*)(ws + 356352);
  float* f2bi = (float*)(ws + 356608);
  float*    attn = (float*)(ws + 356864);        // N*Lq*M*P f32     1048576 B
  _Float16* valh = (_Float16*)(ws + 881152);     // rearranged val 22282240 B
  _Float16* feat = (_Float16*)(ws + 23163392);   // 262144x128 f16 67108864 B
  _Float16* pin  = (_Float16*)(ws + 90272256);   // 4096x256 f16    2097152 B

  // weight transpose/convert + BN fold + attention (independent)
  k_wt_transpose<<<256, 256, 0, stream>>>(vp_w, vp_wt, 256, 256, 65536);
  k_wt_transpose<<<256, 256, 0, stream>>>(op_w, op_wt, 256, 256, 65536);
  k_wt_transpose<<<64,  256, 0, stream>>>(m1_w, m1t, 64, 64, 16384);
  k_wt_transpose<<<32,  256, 0, stream>>>(m2_w, m2t, 64, 32, 8192);
  k_wt_transpose<<<64,  256, 0, stream>>>(f1_w, f1t, 128, 128, 16384);
  k_wt_transpose<<<16,  256, 0, stream>>>(f2_w, f2t, 128, 32, 4096);
  k_bnprep<<<1, 256, 0, stream>>>(m1_g, m1_b, m1_rm, m1_rv, m1sc, m1bi, 256);
  k_bnprep<<<1, 256, 0, stream>>>(m2_g, m2_b, m2_rm, m2_rv, m2sc, m2bi, 128);
  k_bnprep<<<1, 256, 0, stream>>>(f1_g, f1_b, f1_rm, f1_rv, f1sc, f1bi, 128);
  k_bnprep<<<1, 256, 0, stream>>>(f2_g, f2_b, f2_rm, f2_rv, f2sc, f2bi, 32);
  k_attn<<<128, 256, 0, stream>>>(query, aw_w, aw_b, attn);

  // value projection GEMM -> gather-friendly f16 layout
  k_value_proj<<<680, 256, 0, stream>>>(value, vp_wt, vp_b, valh);

  // fused grid-sample + stereo MLP, per level
  k_sample_mlp<<<dim3(4096, LL), 256, 0, stream>>>(
      valh, sloc, m1t, m2t, m1sc, m1bi, m2sc, m2bi, feat);

  // fusion MLP + attention-weighted reduction over points
  k_fusion<<<4096, 256, 0, stream>>>(feat, f1t, f2t, f1sc, f1bi, f2sc, f2bi,
                                     attn, pin);

  // output projection
  k_out_proj<<<64, 256, 0, stream>>>(pin, op_wt, op_b, out);
}
